// RNN_42013370089563
// MI455X (gfx1250) — matrix operand order repaired and checked by reading, not compiled, not run on hardware
//
#include <hip/hip_runtime.h>

// Multilayer RNN scan for MI455X (gfx1250), bf16 WMMA path.
// k0: weight transpose->bf16, bias folding, state init.
// k1: pre0 = x @ Wx0 + (bx0+bh0)  -- fully parallel bf16 WMMA GEMM,
//     stored in WMMA C-fragment layout (one b128 per lane per tile).
// k2: persistent pipelined 2-layer scan, weights staged to LDS once via the
//     Tensor Data Mover (fallback: vector copy), one grid barrier per phase,
//     prefetch of next-step pre0 fragments.

#define SEQT 1024
#define BT   64
#define DT   512
#define HT   512

typedef __bf16 bf16_t;
typedef __attribute__((ext_vector_type(8)))  __bf16 v8bf;
typedef __attribute__((ext_vector_type(16))) __bf16 v16bf;
typedef __attribute__((ext_vector_type(8)))  float  v8f;

#if __has_builtin(__builtin_amdgcn_tanhf)
#define FAST_TANH(x) __builtin_amdgcn_tanhf(x)
#else
#define FAST_TANH(x) tanhf(x)
#endif

// ---------------- workspace layout (bytes), ~70 MB total ----------------
constexpr size_t OFF_PRE0 = 0;                                      // bf16 fragment-layout [SEQ*4*32][32][8]
constexpr size_t OFF_WX0T = OFF_PRE0 + (size_t)SEQT * BT * HT * 2;  // bf16 [H][D] (transposed)
constexpr size_t OFF_WH0T = OFF_WX0T + (size_t)HT * DT * 2;         // bf16 [H][H]
constexpr size_t OFF_WX1T = OFF_WH0T + (size_t)HT * HT * 2;         // bf16 [H][H]
constexpr size_t OFF_WH1T = OFF_WX1T + (size_t)HT * HT * 2;         // bf16 [H][H]
constexpr size_t OFF_B0   = OFF_WH1T + (size_t)HT * HT * 2;         // f32 [H]  (bx0+bh0)
constexpr size_t OFF_B1   = OFF_B0 + (size_t)HT * 4;                // f32 [H]  (bx1+bh1)
constexpr size_t OFF_H0   = OFF_B1 + (size_t)HT * 4;                // bf16 [2][B][H] double buffer
constexpr size_t OFF_H1   = OFF_H0 + (size_t)2 * BT * HT * 2;       // bf16 [2][B][H]
constexpr size_t OFF_SYNC = OFF_H1 + (size_t)2 * BT * HT * 2;       // u32  [64]

// ---------------- WMMA helpers ----------------
__device__ __forceinline__ v8f wmma_bf16(v16bf a, v16bf b, v8f c) {
  return __builtin_amdgcn_wmma_f32_16x16x32_bf16(false, a, false, b, (short)0, c,
                                                 false, false);
}

__device__ __forceinline__ v16bf cat8(v8bf lo, v8bf hi) {
  return __builtin_shufflevector(lo, hi, 0, 1, 2, 3, 4, 5, 6, 7,
                                 8, 9, 10, 11, 12, 13, 14, 15);
}

// A matrix 16x32 bf16 layout: lane<16 -> M=lane, K in {kk..+7, kk+16..+23}
//                             lane>=16 -> M=lane-16, K in {kk+8..+15, kk+24..+31}
__device__ __forceinline__ v16bf load_a_bf16(const bf16_t* __restrict__ A, int lda,
                                             int m, int kk, int lane) {
  int kbase = kk + ((lane & 16) ? 8 : 0);
  const v8bf* p0 = (const v8bf*)(A + (size_t)m * lda + kbase);
  const v8bf* p1 = (const v8bf*)(A + (size_t)m * lda + kbase + 16);
  return cat8(*p0, *p1);
}

// Same A layout, fp32 source (convert in-register).
__device__ __forceinline__ v16bf load_a_f32(const float* __restrict__ A, int lda,
                                            int m, int kk, int lane) {
  int kbase = kk + ((lane & 16) ? 8 : 0);
  const float4* p0 = (const float4*)(A + (size_t)m * lda + kbase);
  const float4* p1 = (const float4*)(A + (size_t)m * lda + kbase + 16);
  float4 f0 = p0[0], f1 = p0[1], f2 = p1[0], f3 = p1[1];
  v16bf r;
  r[0]  = (__bf16)f0.x; r[1]  = (__bf16)f0.y; r[2]  = (__bf16)f0.z; r[3]  = (__bf16)f0.w;
  r[4]  = (__bf16)f1.x; r[5]  = (__bf16)f1.y; r[6]  = (__bf16)f1.z; r[7]  = (__bf16)f1.w;
  r[8]  = (__bf16)f2.x; r[9]  = (__bf16)f2.y; r[10] = (__bf16)f2.z; r[11] = (__bf16)f2.w;
  r[12] = (__bf16)f3.x; r[13] = (__bf16)f3.y; r[14] = (__bf16)f3.z; r[15] = (__bf16)f3.w;
  return r;
}

// B matrix 32x16 bf16 layout: lane<16 -> N=lane,    K in {kk..+7, kk+16..+23}
//                             lane>=16 -> N=lane-16, K in {kk+8..+15, kk+24..+31}
// W is [rows][K] row-major with stride HT; `row` already includes lane column.
__device__ __forceinline__ v16bf load_b_row(const bf16_t* W, int row, int kk, int lane) {
  int koff = kk + ((lane & 16) ? 8 : 0);
  const v8bf* p = (const v8bf*)(W + (size_t)row * HT + koff);
  return cat8(p[0], p[2]);
}

// ---------------- Tensor Data Mover staging (guarded) ----------------
#if __has_builtin(__builtin_amdgcn_tensor_load_to_lds)
#define HAVE_TDM 1
typedef __attribute__((ext_vector_type(4))) unsigned int u32x4;
typedef __attribute__((ext_vector_type(8))) int i32x8;
typedef __attribute__((ext_vector_type(4))) int i32x4;

// Load `rows` consecutive rows of a [rows][512] bf16 tile from global into LDS
// at byte offset lds_off. D# per CDNA5 ISA.
__device__ __forceinline__ void tdm_load_rows_to_lds(const bf16_t* src,
                                                     unsigned lds_off,
                                                     unsigned rows) {
  unsigned long long ga = (unsigned long long)src;
  u32x4 g0 = {
    1u,                                                   // count=1, user desc
    lds_off,                                              // lds_addr (bytes)
    (unsigned)ga,                                         // global_addr[31:0]
    (unsigned)((ga >> 32) & 0x1FFFFFFull) | 0x80000000u   // addr[56:32] | type=2
  };
  i32x8 g1 = {
    (int)(1u << 16),        // data_size = 1 (2 bytes); no mask/iterate/pad
    (int)(512u << 16),      // tensor_dim0[15:0] @ bits 63:48
    (int)(rows << 16),      // tensor_dim0 hi=0 | tensor_dim1[15:0] @ bits 95:80
    (int)(512u << 16),      // tensor_dim1 hi=0 | tile_dim0 @ bits 127:112
    (int)rows,              // tile_dim1 | tile_dim2=0
    512,                    // tensor_dim0_stride lo32
    0, 0                    // stride hi | tensor_dim1_stride = 0
  };
  i32x4 z4 = {0, 0, 0, 0};
#if __has_include(<hip/amd_detail/amd_gfx1250_TDM.h>)
  i32x8 z8 = {0, 0, 0, 0, 0, 0, 0, 0};
  __builtin_amdgcn_tensor_load_to_lds(g0, g1, z4, z4, z8, 0);   // clang-23 form
#else
  __builtin_amdgcn_tensor_load_to_lds(g0, g1, z4, z4, 0);       // ROCm 7.2 form
#endif
}

__device__ __forceinline__ void tdm_wait() {
#if __has_builtin(__builtin_amdgcn_s_wait_tensorcnt)
  __builtin_amdgcn_s_wait_tensorcnt(0);
#else
  asm volatile("s_wait_tensorcnt 0x0" ::: "memory");
#endif
}
#endif  // HAVE_TDM

// ---------------- grid-wide barrier (persistent kernel) ----------------
__device__ __forceinline__ void gridBarrier(unsigned* cnt, unsigned* gen, unsigned s) {
  __syncthreads();
  if (threadIdx.x == 0) {
    unsigned prev = __hip_atomic_fetch_add(cnt, 1u, __ATOMIC_ACQ_REL,
                                           __HIP_MEMORY_SCOPE_AGENT);
    if (prev == gridDim.x - 1u) {
      __hip_atomic_store(cnt, 0u, __ATOMIC_RELAXED, __HIP_MEMORY_SCOPE_AGENT);
      __hip_atomic_store(gen, s + 1u, __ATOMIC_RELEASE, __HIP_MEMORY_SCOPE_AGENT);
    } else {
      while (__hip_atomic_load(gen, __ATOMIC_ACQUIRE, __HIP_MEMORY_SCOPE_AGENT) <= s) {
        __builtin_amdgcn_s_sleep(1);
      }
    }
  }
  __syncthreads();
}

// ---------------- kernel 0: weight transpose/convert + state init ----------------
__global__ void __launch_bounds__(256)
prep_kernel(const float* __restrict__ Wx0, const float* __restrict__ Wh0,
            const float* __restrict__ Wx1, const float* __restrict__ Wh1,
            const float* __restrict__ bx0, const float* __restrict__ bh0,
            const float* __restrict__ bx1, const float* __restrict__ bh1,
            const float* __restrict__ h0, char* __restrict__ ws) {
  int g = blockIdx.x * 256 + threadIdx.x;   // 0 .. 512*512-1
  int d = g >> 9;
  int h = g & 511;
  bf16_t* Wx0t = (bf16_t*)(ws + OFF_WX0T);
  bf16_t* Wh0t = (bf16_t*)(ws + OFF_WH0T);
  bf16_t* Wx1t = (bf16_t*)(ws + OFF_WX1T);
  bf16_t* Wh1t = (bf16_t*)(ws + OFF_WH1T);
  Wx0t[(size_t)h * DT + d] = (__bf16)Wx0[(size_t)d * HT + h];
  Wh0t[(size_t)h * HT + d] = (__bf16)Wh0[(size_t)d * HT + h];
  Wx1t[(size_t)h * HT + d] = (__bf16)Wx1[(size_t)d * HT + h];
  Wh1t[(size_t)h * HT + d] = (__bf16)Wh1[(size_t)d * HT + h];
  if (g < HT) {
    ((float*)(ws + OFF_B0))[g] = bx0[g] + bh0[g];
    ((float*)(ws + OFF_B1))[g] = bx1[g] + bh1[g];
  }
  if (g < BT * HT) {
    ((bf16_t*)(ws + OFF_H0))[g] = (__bf16)h0[g];
    ((bf16_t*)(ws + OFF_H1))[g] = (__bf16)h0[BT * HT + g];
  }
  if (g < 64) ((unsigned*)(ws + OFF_SYNC))[g] = 0u;
}

// ---------------- kernel 1: pre0 = x @ Wx0 + (bx0+bh0) ----------------
// M = SEQ*B = 65536, N = H = 512, K = D = 512. One 16x16 tile per wave.
// Output in WMMA C-fragment layout: pre0[tile_id][lane][8], tile_id = tm*32+tn.
__global__ void __launch_bounds__(256)
precompute_kernel(const float* __restrict__ x, const char* __restrict__ ws_c,
                  char* __restrict__ ws) {
  const bf16_t* Wx0t = (const bf16_t*)(ws_c + OFF_WX0T);
  const float*  b0   = (const float*)(ws_c + OFF_B0);
  bf16_t*       pre0 = (bf16_t*)(ws + OFF_PRE0);

  int lane = threadIdx.x & 31;
  int gw   = blockIdx.x * 8 + (threadIdx.x >> 5);   // 0 .. 131071 tiles
  int tm   = gw >> 5;                               // M tile
  int tn   = gw & 31;                               // N tile
  int m0 = tm * 16, n0 = tn * 16;
  int col = lane & 15;
  int arow = m0 + col;

  v8f acc = {};
#pragma unroll
  for (int kk = 0; kk < DT; kk += 32) {
    v16bf a = load_a_f32(x, DT, arow, kk, lane);
    v16bf b = load_b_row(Wx0t, n0 + col, kk, lane);
    acc = wmma_bf16(a, b, acc);
  }
  float bias = b0[n0 + col];
  v8bf frag;
#pragma unroll
  for (int r = 0; r < 8; ++r) frag[r] = (__bf16)(acc[r] + bias);
  *(v8bf*)(pre0 + ((size_t)gw * 32 + lane) * 8) = frag;   // one b128 store
}

// ---------------- kernel 2: persistent pipelined scan ----------------
// 24 blocks x 256 threads. Weights live in LDS for the whole scan.
//   Blocks 0..7   : layer0. Block owns 4 n-tiles (64 cols) of Wh0t -> 64 KB LDS.
//                   Wave w: n_local = w>>1, m tiles {(w&1)*2, (w&1)*2+1}; B reused x2.
//   Blocks 8..23  : layer1. Block owns 2 n-tiles of Wx1t + Wh1t -> 32+32 KB LDS.
//                   Wave w: n_local = w>>2, m tile = w&3.
// Phase s: layer0 computes t=s (s<SEQ); layer1 computes t=s-1 (s>=1).
__global__ void __launch_bounds__(256)
rnn_scan_kernel(char* __restrict__ ws, float* __restrict__ out) {
  __shared__ bf16_t smem[4 * 16 * HT];   // 64 KB

  const bf16_t* pre0 = (const bf16_t*)(ws + OFF_PRE0);
  const float*  b1   = (const float*)(ws + OFF_B1);
  bf16_t* H0 = (bf16_t*)(ws + OFF_H0);
  bf16_t* H1 = (bf16_t*)(ws + OFF_H1);
  unsigned* cnt = (unsigned*)(ws + OFF_SYNC);
  unsigned* gen = (unsigned*)(ws + OFF_SYNC) + 32;
  float* hn0 = out + (size_t)SEQT * BT * HT;            // h_n layer 0
  float* hn1 = hn0 + (size_t)BT * HT;                   // h_n layer 1

  const int lane  = threadIdx.x & 31;
  const int w     = threadIdx.x >> 5;                   // wave in block: 0..7
  const int col   = lane & 15;
  const int rbase = (lane & 16) ? 8 : 0;
  const bool isL0 = blockIdx.x < 8;

  // ---- stage this block's weight strips into LDS (once for all 1024 steps) ----
#ifdef HAVE_TDM
  if (w == 0) {
    if (isL0) {
      tdm_load_rows_to_lds((const bf16_t*)(ws + OFF_WH0T) + (size_t)blockIdx.x * 64 * HT,
                           0u, 64u);
    } else {
      int bb = blockIdx.x - 8;
      tdm_load_rows_to_lds((const bf16_t*)(ws + OFF_WX1T) + (size_t)bb * 32 * HT,
                           0u, 32u);
      tdm_load_rows_to_lds((const bf16_t*)(ws + OFF_WH1T) + (size_t)bb * 32 * HT,
                           (unsigned)(2 * 16 * HT * sizeof(bf16_t)), 32u);
    }
    tdm_wait();
  }
  // The TDM descriptor references smem only via an integer LDS offset, so the
  // compiler would otherwise treat smem as never-written and DCE the ds_loads.
  // Force smem to escape into the (already issued) DMA:
  asm volatile("" : : "r"(&smem[0]) : "memory");
#else
  if (isL0) {
    const v8bf* src = (const v8bf*)((const bf16_t*)(ws + OFF_WH0T) +
                                    (size_t)blockIdx.x * 64 * HT);
    v8bf* dst = (v8bf*)smem;
    for (int i = threadIdx.x; i < 4096; i += 256) dst[i] = src[i];
  } else {
    int bb = blockIdx.x - 8;
    const v8bf* src0 = (const v8bf*)((const bf16_t*)(ws + OFF_WX1T) +
                                     (size_t)bb * 32 * HT);
    const v8bf* src1 = (const v8bf*)((const bf16_t*)(ws + OFF_WH1T) +
                                     (size_t)bb * 32 * HT);
    v8bf* dst0 = (v8bf*)smem;
    v8bf* dst1 = (v8bf*)(smem + 2 * 16 * HT);
    for (int i = threadIdx.x; i < 2048; i += 256) { dst0[i] = src0[i]; dst1[i] = src1[i]; }
  }
#endif
  __syncthreads();

  if (isL0) {
    const int n_local = w >> 1;                         // 0..3
    const int n0 = blockIdx.x * 64 + n_local * 16;
    const int m0a = (w & 1) * 32;
    const int m0b = m0a + 16;
    const int brow = n_local * 16 + col;                // row into LDS strip
    const int tn = n0 >> 4;
    // fragment tile base ids (per step stride = 4*32 tiles)
    const size_t fa = ((size_t)(m0a >> 4) * 32 + tn) * 256 + (size_t)lane * 8;
    const size_t fb = ((size_t)(m0b >> 4) * 32 + tn) * 256 + (size_t)lane * 8;

    for (int s = 0; s <= SEQT; ++s) {
      if (s < SEQT) {
        int t = s;
        const bf16_t* hprev = H0 + (size_t)(t & 1) * (BT * HT);
        bf16_t*       hnext = H0 + (size_t)((t + 1) & 1) * (BT * HT);
        const bf16_t* prt = pre0 + (size_t)t * (4 * 32 * 256);
        // prefetch next step's pre0 fragments (only HBM stream in the scan)
        if (t + 1 < SEQT) {
          __builtin_prefetch(prt + 4 * 32 * 256 + fa, 0, 1);
          __builtin_prefetch(prt + 4 * 32 * 256 + fb, 0, 1);
        }
        v8f acc0 = {}, acc1 = {};
#pragma unroll
        for (int kk = 0; kk < HT; kk += 32) {
          v16bf b  = load_b_row(smem, brow, kk, lane);          // LDS
          v16bf a0 = load_a_bf16(hprev, HT, m0a + col, kk, lane);
          acc0 = wmma_bf16(a0, b, acc0);
          v16bf a1 = load_a_bf16(hprev, HT, m0b + col, kk, lane);
          acc1 = wmma_bf16(a1, b, acc1);
        }
        v8bf p0 = *(const v8bf*)(prt + fa);                     // one b128 load
        v8bf p1 = *(const v8bf*)(prt + fb);
#pragma unroll
        for (int r = 0; r < 8; ++r) {
          size_t i0 = (size_t)(m0a + rbase + r) * HT + n0 + col;
          size_t i1 = (size_t)(m0b + rbase + r) * HT + n0 + col;
          float v0 = FAST_TANH(acc0[r] + (float)p0[r]);
          float v1 = FAST_TANH(acc1[r] + (float)p1[r]);
          hnext[i0] = (__bf16)v0;
          hnext[i1] = (__bf16)v1;
          if (t == SEQT - 1) { hn0[i0] = v0; hn0[i1] = v1; }
        }
      }
      if (s < SEQT) gridBarrier(cnt, gen, (unsigned)s);
    }
  } else {
    const int bb = blockIdx.x - 8;
    const int n_local = w >> 2;                         // 0..1
    const int n0 = bb * 32 + n_local * 16;
    const int m0 = (w & 3) * 16;
    const int brow = n_local * 16 + col;
    const bf16_t* sWx = smem;                           // Wx1t strip
    const bf16_t* sWh = smem + 2 * 16 * HT;             // Wh1t strip

    for (int s = 0; s <= SEQT; ++s) {
      if (s >= 1) {
        int t = s - 1;
        const bf16_t* h0in  = H0 + (size_t)((t + 1) & 1) * (BT * HT);  // h^0_t
        const bf16_t* h1prv = H1 + (size_t)(t & 1) * (BT * HT);        // h^1_{t-1}
        bf16_t*       h1nxt = H1 + (size_t)((t + 1) & 1) * (BT * HT);
        v8f acc = {};
#pragma unroll
        for (int kk = 0; kk < HT; kk += 32) {
          v16bf bx = load_b_row(sWx, brow, kk, lane);                  // LDS
          v16bf a0 = load_a_bf16(h0in, HT, m0 + col, kk, lane);
          acc = wmma_bf16(a0, bx, acc);
          v16bf bh = load_b_row(sWh, brow, kk, lane);                  // LDS
          v16bf a1 = load_a_bf16(h1prv, HT, m0 + col, kk, lane);
          acc = wmma_bf16(a1, bh, acc);
        }
        float bias = b1[n0 + col];
        float* ot = out + (size_t)t * BT * HT;
#pragma unroll
        for (int r = 0; r < 8; ++r) {
          size_t idx = (size_t)(m0 + rbase + r) * HT + n0 + col;
          float v = FAST_TANH(acc[r] + bias);
          ot[idx] = v;
          h1nxt[idx] = (__bf16)v;
          if (t == SEQT - 1) hn1[idx] = v;
        }
      }
      if (s < SEQT) gridBarrier(cnt, gen, (unsigned)s);
    }
  }
}

// ---------------- launch ----------------
extern "C" void kernel_launch(void* const* d_in, const int* in_sizes, int n_in,
                              void* d_out, int out_size, void* d_ws, size_t ws_size,
                              hipStream_t stream) {
  const float* x   = (const float*)d_in[0];
  const float* h0  = (const float*)d_in[1];
  const float* Wx0 = (const float*)d_in[2];
  const float* bx0 = (const float*)d_in[3];
  const float* Wh0 = (const float*)d_in[4];
  const float* bh0 = (const float*)d_in[5];
  const float* Wx1 = (const float*)d_in[6];
  const float* bx1 = (const float*)d_in[7];
  const float* Wh1 = (const float*)d_in[8];
  const float* bh1 = (const float*)d_in[9];
  float* out = (float*)d_out;
  char*  ws  = (char*)d_ws;

  // 0) transpose/convert weights to bf16, fold biases, init state, reset barrier
  prep_kernel<<<1024, 256, 0, stream>>>(Wx0, Wh0, Wx1, Wh1, bx0, bh0, bx1, bh1, h0, ws);

  // 1) pre0 = x @ Wx0 + b0 : 131072 tiles, 8 waves/block
  precompute_kernel<<<16384, 256, 0, stream>>>(x, ws, ws);

  // 2) persistent pipelined 2-layer scan, weights LDS-resident
  rnn_scan_kernel<<<24, 256, 0, stream>>>(ws, out);
}
